// LSTM4RNMT_22050362098083
// MI455X (gfx1250) — matrix operand order repaired and checked by reading, not compiled, not run on hardware
//
#include <hip/hip_runtime.h>
#include <hip/hip_bf16.h>

#define B_      64
#define T_      512
#define ISZ     1024
#define OSZ     1024
#define KTOT    2048
#define NTOT    4096
#define NWG     64
#define NTHR    256
#define HPAD    (OSZ + 8)          // LDS row stride (shorts), +16B to spread banks
#define LN_EPS  1e-6f

typedef __attribute__((ext_vector_type(16))) __bf16 bf16x16;
typedef __attribute__((ext_vector_type(8)))  float  f32x8;
typedef __attribute__((ext_vector_type(4)))  float  f32x4;

union Frag16 {
    unsigned short us[16];
    int4           i4[2];
    bf16x16        v;
};

__device__ __forceinline__ unsigned short f2bf(float f) {
    unsigned u = __float_as_uint(f);
    u += 0x7FFFu + ((u >> 16) & 1u);       // round-to-nearest-even
    return (unsigned short)(u >> 16);
}

__device__ __forceinline__ f32x8 bfwmma(const Frag16& a, const Frag16& b, f32x8 c) {
    return __builtin_amdgcn_wmma_f32_16x16x32_bf16(false, a.v, false, b.v,
                                                   (short)0, c, false, false);
}

// A-fragment from bf16 row (global or LDS): two b128 loads, per CDNA5 ISA layout
__device__ __forceinline__ void load_a_bf(Frag16& f, const unsigned short* row, int k0) {
    f.i4[0] = *(const int4*)(row + k0);
    f.i4[1] = *(const int4*)(row + k0 + 16);
}

// A-fragment from fp32 global row (fallback MODE 2 only)
__device__ __forceinline__ void load_a_x(Frag16& f, const float* __restrict__ row, int k0) {
    const float4* p = (const float4*)(row + k0);
    float4 a = p[0], b = p[1];             // K = k0 .. k0+7
    float4 c = p[4], d = p[5];             // K = k0+16 .. k0+23
    f.us[0]=f2bf(a.x); f.us[1]=f2bf(a.y); f.us[2]=f2bf(a.z); f.us[3]=f2bf(a.w);
    f.us[4]=f2bf(b.x); f.us[5]=f2bf(b.y); f.us[6]=f2bf(b.z); f.us[7]=f2bf(b.w);
    f.us[8]=f2bf(c.x); f.us[9]=f2bf(c.y); f.us[10]=f2bf(c.z); f.us[11]=f2bf(c.w);
    f.us[12]=f2bf(d.x); f.us[13]=f2bf(d.y); f.us[14]=f2bf(d.z); f.us[15]=f2bf(d.w);
}

__device__ __forceinline__ void grid_barrier(unsigned* bar, unsigned target) {
    __syncthreads();
    if (threadIdx.x == 0) {
        atomicAdd(bar, 1u);
        while (__hip_atomic_load(bar, __ATOMIC_ACQUIRE, __HIP_MEMORY_SCOPE_AGENT) < target) {
            __builtin_amdgcn_s_sleep(2);
        }
    }
    __syncthreads();
}

// ---------------- one-time prep kernels ----------------

__global__ void wt_kernel(const float* __restrict__ W, unsigned short* __restrict__ WT) {
    int idx = blockIdx.x * 256 + threadIdx.x;          // over 4096*2048
    int n = idx >> 11, k = idx & (KTOT - 1);
    WT[(size_t)n * KTOT + k] = f2bf(__builtin_nontemporal_load(&W[(size_t)k * NTOT + n]));
}

__global__ void xbf_kernel(const float* __restrict__ x, unsigned short* __restrict__ xbf) {
    int idx = blockIdx.x * 256 + threadIdx.x;          // float4 index, 8388608 total
    f32x4 v = __builtin_nontemporal_load((const f32x4*)x + idx);
    unsigned long long p = (unsigned long long)f2bf(v[0])
                         | ((unsigned long long)f2bf(v[1]) << 16)
                         | ((unsigned long long)f2bf(v[2]) << 32)
                         | ((unsigned long long)f2bf(v[3]) << 48);
    __builtin_nontemporal_store(p, &((unsigned long long*)xbf)[idx]);
}

__global__ void init_kernel(const float* __restrict__ hx, const float* __restrict__ cx,
                            unsigned short* __restrict__ h_state, float* __restrict__ c_state,
                            float* __restrict__ rowstats, unsigned* __restrict__ bar) {
    int idx = blockIdx.x * 256 + threadIdx.x;          // 64*1024
    int o = idx & (OSZ - 1);
    h_state[idx] = f2bf(hx[o]);
    c_state[idx] = cx[o];
    if (idx < 2 * B_ * 2) rowstats[idx] = 0.0f;
    if (idx == 0) *bar = 0u;
}

// ---------------- parallel precompute GEMM: Xg = x @ Wx + b ----------------
// 128x64 block, 8 wave32s, 4 tiles/wave (2 A-frags + 2 B-frags -> 4 WMMAs)

__global__ __launch_bounds__(NTHR)
void pre_gemm(const unsigned short* __restrict__ xbf, const unsigned short* __restrict__ WT,
              const float* __restrict__ bias, float* __restrict__ Xg) {
    const int tid  = threadIdx.x;
    const int lane = tid & 31, wave = tid >> 5;
    const int half = lane >> 4, l16 = lane & 15;
    const int mblk = blockIdx.x >> 6;                  // 256 M-blocks of 128 rows
    const int nblk = blockIdx.x & 63;                  // 64  N-blocks of 64 cols
    const int mt0  = 2 * (wave & 3);                   // M-tiles mt0, mt0+1 (of 8)
    const int nt0  = 2 * (wave >> 2);                  // N-tiles nt0, nt0+1 (of 4)

    const int rowA0 = mblk * 128 + mt0 * 16 + l16;
    const int rowA1 = rowA0 + 16;
    const int ncol0 = nblk * 64 + nt0 * 16 + l16;
    const int ncol1 = ncol0 + 16;
    const unsigned short* xr0 = xbf + (size_t)rowA0 * ISZ;
    const unsigned short* xr1 = xbf + (size_t)rowA1 * ISZ;
    const unsigned short* wr0 = WT + (size_t)ncol0 * KTOT;
    const unsigned short* wr1 = WT + (size_t)ncol1 * KTOT;

    f32x8 a00 = {0.f,0.f,0.f,0.f,0.f,0.f,0.f,0.f};
    f32x8 a01 = a00, a10 = a00, a11 = a00;

    for (int kk = 0; kk < ISZ; kk += 32) {
        Frag16 fa0, fa1, fb0, fb1;
        const int k0 = kk + 8 * half;
        load_a_bf(fa0, xr0, k0);
        load_a_bf(fa1, xr1, k0);
        const int kb = kk + 8 * half;
        fb0.i4[0] = *(const int4*)(wr0 + kb); fb0.i4[1] = *(const int4*)(wr0 + kb + 16);
        fb1.i4[0] = *(const int4*)(wr1 + kb); fb1.i4[1] = *(const int4*)(wr1 + kb + 16);
        a00 = bfwmma(fa0, fb0, a00);
        a01 = bfwmma(fa0, fb1, a01);
        a10 = bfwmma(fa1, fb0, a10);
        a11 = bfwmma(fa1, fb1, a11);
    }
    const float bn0 = bias[ncol0], bn1 = bias[ncol1];
    #pragma unroll
    for (int r = 0; r < 8; ++r) {
        int ra = mblk * 128 + mt0 * 16 + 8 * half + r;
        int rb = ra + 16;
        __builtin_nontemporal_store(a00[r] + bn0, &Xg[(size_t)ra * NTOT + ncol0]);
        __builtin_nontemporal_store(a01[r] + bn1, &Xg[(size_t)ra * NTOT + ncol1]);
        __builtin_nontemporal_store(a10[r] + bn0, &Xg[(size_t)rb * NTOT + ncol0]);
        __builtin_nontemporal_store(a11[r] + bn1, &Xg[(size_t)rb * NTOT + ncol1]);
    }
}

// ---------------- persistent LSTM scan ----------------
// MODE 0: serial K=1024 (h only), comb += precomputed Xg (includes bias)
// MODE 1: serial K=2048, x-half fragments from bf16 xbf
// MODE 2: serial K=2048, x-half fragments converted from fp32 in-loop

template <int MODE>
__global__ __launch_bounds__(NTHR, 1)
void lstm_kernel(const float* __restrict__ x, const unsigned short* __restrict__ xbf,
                 const float* __restrict__ Xg, const unsigned short* __restrict__ WT,
                 const float* __restrict__ bias, const float* __restrict__ lng,
                 const float* __restrict__ lnb,
                 unsigned short* __restrict__ h_state, float* __restrict__ c_state,
                 float* __restrict__ rowstats, unsigned* __restrict__ bar,
                 float* __restrict__ out) {
    __shared__ __attribute__((aligned(16))) unsigned short hlds[B_ * HPAD]; // 132 KB
    __shared__ float comb[4][B_][16];                                       // 16 KB

    const int tid  = threadIdx.x;
    const int wg   = blockIdx.x;
    const int lane = tid & 31;
    const int wave = tid >> 5;
    const int half = lane >> 4;
    const int l16  = lane & 15;
    const int gate = wave >> 1;        // waves 0..7 -> gate 0..3
    const int mpair= wave & 1;         // which pair of M-tiles

    const int ncol = gate * OSZ + wg * 16 + l16;            // W column for this lane
    const float bias_n = bias[ncol];
    const unsigned short* wtrow = WT + (size_t)ncol * KTOT; // transposed weight row

    const int m0 = (2 * mpair) * 16;
    const int m1 = m0 + 16;
    const int rowA0 = m0 + l16;        // A rows (batch index), same for both halves
    const int rowA1 = m1 + l16;
    const unsigned short* hrow0 = hlds + rowA0 * HPAD;
    const unsigned short* hrow1 = hlds + rowA1 * HPAD;

    unsigned epoch = 0;

    for (int t = 0; t < T_; ++t) {
        // stage full h[64][1024] (bf16) into LDS, 16B chunks
        for (int idx = tid; idx < B_ * (OSZ / 8); idx += NTHR) {
            int r = idx >> 7, cq = idx & 127;
            *(int4*)(&hlds[r * HPAD + cq * 8]) =
                *(const int4*)(h_state + (size_t)r * OSZ + cq * 8);
        }

        // MODE 0: issue the streamed Xg loads NOW so their latency overlaps
        // the whole WMMA loop (they are only consumed at accumulator spill).
        float xg0[8], xg1[8];
        if constexpr (MODE == 0) {
            #pragma unroll
            for (int r = 0; r < 8; ++r) {
                int ba = m0 + 8 * half + r, bb = m1 + 8 * half + r;
                xg0[r] = __builtin_nontemporal_load(&Xg[((size_t)ba * T_ + t) * NTOT + ncol]);
                xg1[r] = __builtin_nontemporal_load(&Xg[((size_t)bb * T_ + t) * NTOT + ncol]);
            }
        }
        __syncthreads();

        f32x8 acc0 = {0.f,0.f,0.f,0.f,0.f,0.f,0.f,0.f};
        f32x8 acc1 = {0.f,0.f,0.f,0.f,0.f,0.f,0.f,0.f};

        // K = 0..1023 : x half of icat (skipped in MODE 0 — folded into Xg)
        if constexpr (MODE == 1) {
            const unsigned short* xr0 = xbf + ((size_t)rowA0 * T_ + t) * ISZ;
            const unsigned short* xr1 = xbf + ((size_t)rowA1 * T_ + t) * ISZ;
            for (int kk = 0; kk < ISZ; kk += 32) {
                Frag16 fb, fa0, fa1;
                const unsigned short* bp = wtrow + kk + 8 * half;
                fb.i4[0] = *(const int4*)(bp);
                fb.i4[1] = *(const int4*)(bp + 16);
                const int k0 = kk + 8 * half;
                load_a_bf(fa0, xr0, k0);
                load_a_bf(fa1, xr1, k0);
                acc0 = bfwmma(fa0, fb, acc0);
                acc1 = bfwmma(fa1, fb, acc1);
            }
        } else if constexpr (MODE == 2) {
            const float* xr0 = x + ((size_t)rowA0 * T_ + t) * ISZ;
            const float* xr1 = x + ((size_t)rowA1 * T_ + t) * ISZ;
            for (int kk = 0; kk < ISZ; kk += 32) {
                Frag16 fb, fa0, fa1;
                const unsigned short* bp = wtrow + kk + 8 * half;
                fb.i4[0] = *(const int4*)(bp);
                fb.i4[1] = *(const int4*)(bp + 16);
                const int k0 = kk + 8 * half;
                load_a_x(fa0, xr0, k0);
                load_a_x(fa1, xr1, k0);
                acc0 = bfwmma(fa0, fb, acc0);
                acc1 = bfwmma(fa1, fb, acc1);
            }
        }

        // K = 1024..2047 : h half of icat (bf16 LDS fragments)
        for (int kk = ISZ; kk < KTOT; kk += 32) {
            Frag16 fb, fa0, fa1;
            const unsigned short* bp = wtrow + kk + 8 * half;
            fb.i4[0] = *(const int4*)(bp);
            fb.i4[1] = *(const int4*)(bp + 16);
            const int k0 = (kk - ISZ) + 8 * half;
            load_a_bf(fa0, hrow0, k0);
            load_a_bf(fa1, hrow1, k0);
            acc0 = bfwmma(fa0, fb, acc0);
            acc1 = bfwmma(fa1, fb, acc1);
        }

        // D layout: vgpr r -> row = mbase + 8*half + r, col = l16
        if constexpr (MODE == 0) {
            #pragma unroll
            for (int r = 0; r < 8; ++r) {
                comb[gate][m0 + 8 * half + r][l16] = acc0[r] + xg0[r];
                comb[gate][m1 + 8 * half + r][l16] = acc1[r] + xg1[r];
            }
        } else {
            #pragma unroll
            for (int r = 0; r < 8; ++r) {
                comb[gate][m0 + 8 * half + r][l16] = acc0[r] + bias_n;
                comb[gate][m1 + 8 * half + r][l16] = acc1[r] + bias_n;
            }
        }
        __syncthreads();

        // partial row sums for joint LayerNorm over (4, O)
        if (tid < B_) {
            float s = 0.f, s2 = 0.f;
            #pragma unroll
            for (int g = 0; g < 4; ++g)
                for (int c = 0; c < 16; ++c) {
                    float v = comb[g][tid][c];
                    s += v; s2 += v * v;
                }
            float* rs = rowstats + (size_t)(t & 1) * B_ * 2 + tid * 2;
            atomicAdd(&rs[0], s);
            atomicAdd(&rs[1], s2);
        }
        __threadfence();
        grid_barrier(bar, (++epoch) * NWG);

        // epilogue: normalize, gates, c/h update for this WG's 16 output columns
        const float* rs = rowstats + (size_t)(t & 1) * B_ * 2;
        for (int i = tid; i < B_ * 16; i += NTHR) {
            int b = i >> 4, c = i & 15;
            int o = wg * 16 + c;
            float mean = rs[b * 2]     * (1.0f / 4096.0f);
            float var  = rs[b * 2 + 1] * (1.0f / 4096.0f) - mean * mean;
            float inv  = rsqrtf(var + LN_EPS);
            float v0 = (comb[0][b][c] - mean) * inv * lng[0 * OSZ + o] + lnb[0 * OSZ + o];
            float v1 = (comb[1][b][c] - mean) * inv * lng[1 * OSZ + o] + lnb[1 * OSZ + o];
            float v2 = (comb[2][b][c] - mean) * inv * lng[2 * OSZ + o] + lnb[2 * OSZ + o];
            float v3 = (comb[3][b][c] - mean) * inv * lng[3 * OSZ + o] + lnb[3 * OSZ + o];
            float ig = 1.0f / (1.0f + __expf(-v0));
            float fg = 1.0f / (1.0f + __expf(-v1));
            float og = 1.0f / (1.0f + __expf(-v2));
            float hid = tanhf(v3);
            size_t so = (size_t)b * OSZ + o;
            float cn = fg * c_state[so] + ig * hid;
            float hn = og * cn;                     // RNMT: no tanh on cell
            c_state[so] = cn;
            h_state[so] = f2bf(hn);
            __builtin_nontemporal_store(hn, &out[((size_t)b * T_ + t) * OSZ + o]);
        }
        // reset next-parity stats for step t+1
        if (wg == 0 && tid < B_ * 2) rowstats[(size_t)(1 - (t & 1)) * B_ * 2 + tid] = 0.0f;

        __threadfence();
        grid_barrier(bar, (++epoch) * NWG);
    }
}

extern "C" void kernel_launch(void* const* d_in, const int* in_sizes, int n_in,
                              void* d_out, int out_size, void* d_ws, size_t ws_size,
                              hipStream_t stream) {
    const float* x    = (const float*)d_in[0];   // [64,512,1024]
    const float* W    = (const float*)d_in[1];   // [2048,4096]
    const float* bias = (const float*)d_in[2];   // [4096]
    const float* lng  = (const float*)d_in[3];   // [4,1024]
    const float* lnb  = (const float*)d_in[4];   // [4,1024]
    const float* hx   = (const float*)d_in[5];   // [1,1024]
    const float* cx   = (const float*)d_in[6];   // [1,1024]
    float* out = (float*)d_out;

    // workspace layout
    const size_t OFF_WT  = 0;                          // 16 MB  bf16 W^T [4096][2048]
    const size_t OFF_H   = OFF_WT  + 16777216;         // 128 KB bf16 h state
    const size_t OFF_C   = OFF_H   + 131072;           // 256 KB f32 c state
    const size_t OFF_RS  = OFF_C   + 262144;           // 1 KB   LN row stats (x2 parity)
    const size_t OFF_BAR = OFF_RS  + 1024;             // barrier counter (padded)
    const size_t OFF_XBF = OFF_BAR + 1024;             // 64 MB  bf16 x
    const size_t OFF_XG  = OFF_XBF + 67108864;         // 512 MB f32 Xg = x@Wx + b
    const size_t NEED0   = OFF_XG  + 536870912;
    const size_t NEED1   = OFF_XG;

    char* ws = (char*)d_ws;
    unsigned short* WT      = (unsigned short*)(ws + OFF_WT);
    unsigned short* h_state = (unsigned short*)(ws + OFF_H);
    float*          c_state = (float*)(ws + OFF_C);
    float*          rstats  = (float*)(ws + OFF_RS);
    unsigned*       bar     = (unsigned*)(ws + OFF_BAR);
    unsigned short* xbf     = (unsigned short*)(ws + OFF_XBF);
    float*          Xg      = (float*)(ws + OFF_XG);

    const int mode = (ws_size >= NEED0) ? 0 : (ws_size >= NEED1) ? 1 : 2;

    wt_kernel  <<<(NTOT * KTOT) / 256, 256, 0, stream>>>(W, WT);
    init_kernel<<<(B_ * OSZ) / 256,    256, 0, stream>>>(hx, cx, h_state, c_state, rstats, bar);
    if (mode <= 1)
        xbf_kernel<<<(B_ * T_ * ISZ / 4) / 256, 256, 0, stream>>>(x, xbf);
    if (mode == 0)
        pre_gemm<<<(B_ * T_ / 128) * (NTOT / 64), NTHR, 0, stream>>>(xbf, WT, bias, Xg);

    if (mode == 0)
        lstm_kernel<0><<<NWG, NTHR, 0, stream>>>(x, xbf, Xg, WT, bias, lng, lnb,
                                                 h_state, c_state, rstats, bar, out);
    else if (mode == 1)
        lstm_kernel<1><<<NWG, NTHR, 0, stream>>>(x, xbf, Xg, WT, bias, lng, lnb,
                                                 h_state, c_state, rstats, bar, out);
    else
        lstm_kernel<2><<<NWG, NTHR, 0, stream>>>(x, xbf, Xg, WT, bias, lng, lnb,
                                                 h_state, c_state, rstats, bar, out);
}
